// TasteGNN_25658134626980
// MI455X (gfx1250) — matrix operand order repaired and checked by reading, not compiled, not run on hardware
//
#include <hip/hip_runtime.h>
#include <hip/hip_bf16.h>

#define D_DIM 128
#define NEG_SLOPE 0.2f
#define R_ALPHA 0.5f

typedef __attribute__((ext_vector_type(16))) __bf16 v16bf;
typedef __attribute__((ext_vector_type(8)))  float  v8f;

union FragBF { v16bf v; uint4 q[2]; unsigned short u[16]; };

__device__ __forceinline__ unsigned short f2bf(float f) {
  unsigned u = __float_as_uint(f);
  unsigned r = ((u >> 16) & 1u) + 0x7FFFu;   // round-to-nearest-even
  return (unsigned short)((u + r) >> 16);
}

// order-preserving float<->uint encoding (for atomicMax on signed floats)
__device__ __forceinline__ unsigned fenc(float f) {
  unsigned u = __float_as_uint(f);
  return (u & 0x80000000u) ? ~u : (u | 0x80000000u);
}
__device__ __forceinline__ float fdec(unsigned k) {
  unsigned u = (k & 0x80000000u) ? (k & 0x7FFFFFFFu) : ~k;
  return __uint_as_float(u);
}

// Guaranteed-native f32 scatter add: GLOBAL_ATOMIC_ADD_F32 (no return ->
// STOREcnt), device scope so L2 performs the RMW coherently across WGPs.
__device__ __forceinline__ void atomic_add_f32(float* p, float v) {
  asm volatile("global_atomic_add_f32 %0, %1, off scope:SCOPE_DEV"
               :: "v"(p), "v"(v) : "memory");
}

// ---------------------------------------------------------------------------
// H = X @ W + bias   (M x 128) = (M x 128)(128 x 128), bf16 WMMA, fp32 accum.
// One wave computes a 16x128 output stripe (8 N-tiles x 4 K-steps of
// v_wmma_f32_16x16x32_bf16). Fragment layouts (ISA 7.12.2):
//   A (16x32 bf16): lane -> M=lane&15; elem e -> K = ks*32 + hi*8 + e      (e<8)
//                                            K = ks*32 + hi*8 + 16 + e-8  (e>=8)
//     -> two contiguous 16B runs in row-major sA[m][k]  => 2x ds_load_b128
//   B (32x16 bf16): lane -> N=lane&15; elem e -> same K as A
//     -> two contiguous 16B runs in K; store W TRANSPOSED sWt[N][K]
//        => 2x ds_load_b128
//   C/D (16x16 f32): vgpr r -> M = r + hi*8, N = lane&15
// ---------------------------------------------------------------------------
__global__ __launch_bounds__(256) void gemm_bias_wmma(
    const float* __restrict__ X, const float* __restrict__ W,
    const float* __restrict__ bias, float* __restrict__ H, int M)
{
  __shared__ unsigned short sWt[D_DIM][D_DIM + 8];  // [N][K] bf16, 272B stride
  __shared__ unsigned short sA[8][16][D_DIM + 8];   // per-wave [m][k] bf16
  __shared__ float sBias[D_DIM];

  const int tid  = threadIdx.x;
  const int wave = tid >> 5;
  const int lane = tid & 31;

  // cooperative load + convert + transpose of W (coalesced global reads)
  for (int i = 0; i < 64; ++i) {
    int idx = i * 256 + tid;                 // idx = k*128 + n
    sWt[idx & 127][idx >> 7] = f2bf(W[idx]); // sWt[n][k] = W[k][n]
  }
  if (tid < D_DIM) sBias[tid] = bias[tid];
  __syncthreads();

  int m0 = (blockIdx.x * 8 + wave) * 16;
  if (m0 > M - 16) m0 = M - 16;   // clamp; M is a multiple of 16 -> dup writes

  // stage A tile (16 rows x 128 cols) as bf16, coalesced float4 loads
  for (int r = 0; r < 16; ++r) {
    const float4 a4 = ((const float4*)(X + (size_t)(m0 + r) * D_DIM))[lane];
    int c = lane * 4;
    sA[wave][r][c + 0] = f2bf(a4.x);
    sA[wave][r][c + 1] = f2bf(a4.y);
    sA[wave][r][c + 2] = f2bf(a4.z);
    sA[wave][r][c + 3] = f2bf(a4.w);
  }
  __syncthreads();

  v8f acc[8] = {};
  const int nl = lane & 15;
  const int hi = (lane >> 4) & 1;

  for (int ks = 0; ks < 4; ++ks) {
    FragBF a;
    {
      const unsigned short* ap = &sA[wave][nl][ks * 32 + hi * 8];
      a.q[0] = *(const uint4*)(ap);        // elems 0..7   (K .. K+7)
      a.q[1] = *(const uint4*)(ap + 16);   // elems 8..15  (K+16 .. K+23)
    }
#pragma unroll
    for (int nt = 0; nt < 8; ++nt) {
      FragBF b;
      const unsigned short* bp = &sWt[nt * 16 + nl][ks * 32 + hi * 8];
      b.q[0] = ((const uint4*)bp)[0];      // elems 0..7   (K .. K+7)
      b.q[1] = ((const uint4*)bp)[2];      // elems 8..15  (K+16 .. K+23)
      acc[nt] = __builtin_amdgcn_wmma_f32_16x16x32_bf16(
          false, a.v, false, b.v, (short)0, acc[nt], false, false);
    }
  }

#pragma unroll
  for (int nt = 0; nt < 8; ++nt) {
#pragma unroll
    for (int r = 0; r < 8; ++r) {
      int row = m0 + r + hi * 8;
      int col = nt * 16 + nl;
      H[(size_t)row * D_DIM + col] = acc[nt][r] + sBias[col];
    }
  }
}

// ---------------------------------------------------------------------------
// wv[k] = sum_j W_taste[k][j]*att_dst[j];  c0 = sum_j b_taste[j]*att_dst[j]
// (folds the h_taste GEMM: a_dst = x_taste @ wv + c0)
// ---------------------------------------------------------------------------
__global__ void prep_wv(const float* __restrict__ Wt, const float* __restrict__ bt,
                        const float* __restrict__ ad, float* __restrict__ wv,
                        float* __restrict__ c0)
{
  int k = threadIdx.x;   // 128 threads
  float acc = 0.f;
  for (int j = 0; j < D_DIM; ++j) acc += Wt[k * D_DIM + j] * ad[j];
  wv[k] = acc;
  if (k == 0) {
    float c = 0.f;
    for (int j = 0; j < D_DIM; ++j) c += bt[j] * ad[j];
    *c0 = c;
  }
}

// out[i] = dot(X[i,:], vec) + (*biasp or 0) ; one wave per row
__global__ __launch_bounds__(256) void row_dot(
    const float* __restrict__ X, const float* __restrict__ vec,
    const float* __restrict__ biasp, float* __restrict__ out, int N)
{
  int g = blockIdx.x * blockDim.x + threadIdx.x;
  int row = g >> 5, lane = g & 31;
  if (row >= N) return;
  const float4 x4 = ((const float4*)(X + (size_t)row * D_DIM))[lane];
  const float4 v4 = ((const float4*)vec)[lane];
  float p = x4.x * v4.x + x4.y * v4.y + x4.z * v4.z + x4.w * v4.w;
  for (int off = 16; off > 0; off >>= 1) p += __shfl_down(p, off, 32);
  if (lane == 0) out[row] = p + (biasp ? biasp[0] : 0.0f);
}

__global__ void init_ms(unsigned* __restrict__ menc, float* __restrict__ s, int N)
{
  int i = blockIdx.x * blockDim.x + threadIdx.x;
  if (i < N) { menc[i] = 0x007FFFFFu; /* fenc(-inf) */ s[i] = 0.f; }
}

// per-edge leaky_relu logit + segment max (ordered-uint atomicMax over dst)
__global__ void edge_logit_max(const float* __restrict__ a_src, const float* __restrict__ a_dst,
                               const int* __restrict__ src, const int* __restrict__ dst,
                               float* __restrict__ elog, unsigned* __restrict__ menc, int E)
{
  int e = blockIdx.x * blockDim.x + threadIdx.x;
  if (e >= E) return;
  float l = a_src[src[e]] + a_dst[dst[e]];
  l = (l > 0.f) ? l : NEG_SLOPE * l;
  elog[e] = l;
  atomicMax(&menc[dst[e]], fenc(l));
}

// per-edge exp(l - max) + segment sum
__global__ void edge_exp_sum(const float* __restrict__ elog, const int* __restrict__ dst,
                             const unsigned* __restrict__ menc, float* __restrict__ eexp,
                             float* __restrict__ s, int E)
{
  int e = blockIdx.x * blockDim.x + threadIdx.x;
  if (e >= E) return;
  int d = dst[e];
  float x = __expf(elog[e] - fdec(menc[d]));
  eexp[e] = x;
  atomic_add_f32(&s[d], x);
}

// one wave per edge: agg[dst] += (eexp/s[dst]) * h_ing[src]   (128 f32)
__global__ __launch_bounds__(256) void edge_aggregate(
    const float* __restrict__ h, const float* __restrict__ eexp,
    const float* __restrict__ s, const int* __restrict__ src,
    const int* __restrict__ dst, float* __restrict__ agg, int E)
{
  int g = blockIdx.x * blockDim.x + threadIdx.x;
  int e = g >> 5, lane = g & 31;
  if (e >= E) return;
  int si = src[e], d = dst[e];
  float w = eexp[e] / (s[d] + 1e-16f);
  const float4 v = ((const float4*)(h + (size_t)si * D_DIM))[lane];
  float* base = agg + (size_t)d * D_DIM + lane * 4;
  atomic_add_f32(base + 0, v.x * w);
  atomic_add_f32(base + 1, v.y * w);
  atomic_add_f32(base + 2, v.z * w);
  atomic_add_f32(base + 3, v.w * w);
}

// d_out[: n_ing4] = x_ing (passthrough), d_out[n_ing4 : total4] = 0 (agg init)
__global__ void init_out(const float* __restrict__ x_ing, float* __restrict__ out,
                         long n_ing4, long total4)
{
  long i = (long)blockIdx.x * blockDim.x + threadIdx.x;
  long stride = (long)gridDim.x * blockDim.x;
  const float4* xi = (const float4*)x_ing;
  float4* o = (float4*)out;
  for (; i < total4; i += stride)
    o[i] = (i < n_ing4) ? xi[i] : make_float4(0.f, 0.f, 0.f, 0.f);
}

// out_t = relu(out_t) * (1-R_ALPHA) + R_ALPHA * x_taste   (attn == 1 exactly)
__global__ void finalize_taste(const float* __restrict__ x_taste,
                               float* __restrict__ out_t, long n4)
{
  long i = (long)blockIdx.x * blockDim.x + threadIdx.x;
  long stride = (long)gridDim.x * blockDim.x;
  const float4* xt = (const float4*)x_taste;
  float4* o = (float4*)out_t;
  for (; i < n4; i += stride) {
    float4 v = o[i], x = xt[i];
    v.x = fmaxf(v.x, 0.f) * (1.f - R_ALPHA) + R_ALPHA * x.x;
    v.y = fmaxf(v.y, 0.f) * (1.f - R_ALPHA) + R_ALPHA * x.y;
    v.z = fmaxf(v.z, 0.f) * (1.f - R_ALPHA) + R_ALPHA * x.z;
    v.w = fmaxf(v.w, 0.f) * (1.f - R_ALPHA) + R_ALPHA * x.w;
    o[i] = v;
  }
}

extern "C" void kernel_launch(void* const* d_in, const int* in_sizes, int n_in,
                              void* d_out, int out_size, void* d_ws, size_t ws_size,
                              hipStream_t stream)
{
  (void)n_in; (void)out_size; (void)ws_size;
  const float* x_ing   = (const float*)d_in[0];
  const float* x_taste = (const float*)d_in[1];
  const float* W_ing   = (const float*)d_in[2];
  const float* b_ing   = (const float*)d_in[3];
  const float* W_taste = (const float*)d_in[4];
  const float* b_taste = (const float*)d_in[5];
  const float* att_src = (const float*)d_in[6];
  const float* att_dst = (const float*)d_in[7];
  // d_in[8..10] = Wk, bk, q: dead (softmax over 1 element == 1)
  const int* src_idx = (const int*)d_in[11];
  const int* dst_idx = (const int*)d_in[12];

  const int N_ING   = in_sizes[0] / D_DIM;
  const int N_TASTE = in_sizes[1] / D_DIM;
  const int E       = in_sizes[11];

  float* out       = (float*)d_out;
  float* out_taste = out + (size_t)N_ING * D_DIM;

  // workspace partition
  float*    h_ing = (float*)d_ws;                         // N_ING*128
  float*    a_src = h_ing + (size_t)N_ING * D_DIM;        // N_ING
  float*    a_dst = a_src + N_ING;                        // N_TASTE
  unsigned* menc  = (unsigned*)(a_dst + N_TASTE);         // N_TASTE
  float*    s_sum = (float*)(menc + N_TASTE);             // N_TASTE
  float*    wv    = s_sum + N_TASTE;                      // 128
  float*    c0    = wv + D_DIM;                           // 1
  float*    elog  = c0 + 1;                               // E
  float*    eexp  = elog + E;                             // E

  const long n_ing4 = (long)N_ING * D_DIM / 4;
  const long total4 = (long)(N_ING + N_TASTE) * D_DIM / 4;

  init_out<<<2048, 256, 0, stream>>>(x_ing, out, n_ing4, total4);
  prep_wv<<<1, 128, 0, stream>>>(W_taste, b_taste, att_dst, wv, c0);

  const int mtiles = (N_ING + 15) / 16;
  gemm_bias_wmma<<<(mtiles + 7) / 8, 256, 0, stream>>>(x_ing, W_ing, b_ing, h_ing, N_ING);

  row_dot<<<(N_ING * 32 + 255) / 256, 256, 0, stream>>>(h_ing, att_src, nullptr, a_src, N_ING);
  row_dot<<<(N_TASTE * 32 + 255) / 256, 256, 0, stream>>>(x_taste, wv, c0, a_dst, N_TASTE);

  init_ms<<<(N_TASTE + 255) / 256, 256, 0, stream>>>(menc, s_sum, N_TASTE);
  edge_logit_max<<<(E + 255) / 256, 256, 0, stream>>>(a_src, a_dst, src_idx, dst_idx, elog, menc, E);
  edge_exp_sum<<<(E + 255) / 256, 256, 0, stream>>>(elog, dst_idx, menc, eexp, s_sum, E);
  edge_aggregate<<<(E + 7) / 8, 256, 0, stream>>>(h_ing, eexp, s_sum, src_idx, dst_idx, out_taste, E);

  finalize_taste<<<((long)N_TASTE * D_DIM / 4 + 255) / 256, 256, 0, stream>>>(
      x_taste, out_taste, (long)N_TASTE * D_DIM / 4);
}
